// LSTMModel_6193342841502
// MI455X (gfx1250) — compile-verified, matrix operand order checked by reading, not run on hardware
//
#include <hip/hip_runtime.h>
#include <hip/hip_bf16.h>

// ---------- types ----------
typedef __attribute__((ext_vector_type(16))) __bf16 v16bf;
typedef __attribute__((ext_vector_type(8)))  float  v8f;
typedef unsigned short u16;

#define T_SEQ   512
#define B_SZ    16
#define E_DIM   1024
#define H_DIM   1024
#define G4H     4096
#define VOUT    32000
#define M_ROWS  (T_SEQ * B_SZ)     // 8192
#define KT_H    (H_DIM / 32)       // 32 k-tiles per 1024-K

// fp32 -> bf16 (round-to-nearest-even)
__device__ __forceinline__ u16 f2bf(float x) {
  unsigned u = __float_as_uint(x);
  u += 0x7FFFu + ((u >> 16) & 1u);
  return (u16)(u >> 16);
}

// ---------------------------------------------------------------------------
// Fragment layout conventions (per CDNA5 ISA 7.12.2, wave32):
//  A-frag 16x32 bf16 (MxK): lane l holds row M=(l&15); slot i (0..15) holds
//    K = (i&7) + 8*(l>>4) + 16*(i>>3).  Stored contiguously: frag*512 + l*16 + i.
//  B-frag 32x16 bf16 (KxN): lane l holds col N=(l&15); same K(slot,l) mapping.
//  C-frag 16x16 f32: VGPR r, lane l -> M = r + 8*(l>>4), N = (l&15).
// ---------------------------------------------------------------------------

// Zero the grid-barrier counters (ws is poisoned 0xAA by harness).
__global__ void init_counters(unsigned* ctr) {
  if (threadIdx.x < 64) ctr[threadIdx.x] = 0u;
}

// Pack fp32 weight [Nn x Kk] row-major into bf16 B-frag tiles.
// perm==1: interleave gates: packed col n corresponds to source row
//          (n&3)*1024 + (n>>2)   (requires Nn == 4096).
__global__ void pack_w(const float* __restrict__ W, u16* __restrict__ out,
                       int Nn, int Kk, int perm) {
  size_t id = (size_t)blockIdx.x * 256 + threadIdx.x;
  size_t total = (size_t)Nn * Kk;
  if (id >= total) return;
  int slot = (int)(id & 15);
  int l    = (int)((id >> 4) & 31);
  size_t frag = id >> 9;
  int KT = Kk >> 5;
  int kt = (int)(frag % KT);
  int nt = (int)(frag / KT);
  int n  = nt * 16 + (l & 15);
  int k  = kt * 32 + ((slot & 7) + 8 * (l >> 4) + 16 * (slot >> 3));
  int orow = perm ? (((n & 3) << 10) + (n >> 2)) : n;
  out[id] = f2bf(W[(size_t)orow * Kk + k]);
}

// Embedding gather -> bf16 A-frag tiles, rows m = t*16 + b  ([T,B,E] order).
__global__ void pack_embed(const int* __restrict__ x, const float* __restrict__ emb,
                           u16* __restrict__ out) {
  size_t id = (size_t)blockIdx.x * 256 + threadIdx.x;     // one per element
  if (id >= (size_t)M_ROWS * E_DIM) return;
  int slot = (int)(id & 15);
  int l    = (int)((id >> 4) & 31);
  size_t frag = id >> 9;
  int kt = (int)(frag % KT_H);
  int mt = (int)(frag / KT_H);
  int m  = mt * 16 + (l & 15);
  int t  = m >> 4;
  int b  = m & 15;
  int k  = kt * 32 + ((slot & 7) + 8 * (l >> 4) + 16 * (slot >> 3));
  int tok = x[b * T_SEQ + t];
  out[id] = f2bf(emb[(size_t)tok * E_DIM + k]);
}

// ---------------------------------------------------------------------------
// Async-to-LDS helpers (CDNA5 async path, ISA ch.10.7 / 15.18).
// LDS destination address = flat address truncated to [31:0] (aperture rule).
// ---------------------------------------------------------------------------
__device__ __forceinline__ void async_ld16(void* lds_ptr, const void* gptr) {
  unsigned lds_addr = (unsigned)(uintptr_t)lds_ptr;
  unsigned long long ga = (unsigned long long)(uintptr_t)gptr;
  asm volatile("global_load_async_to_lds_b128 %0, %1, off"
               :: "v"(lds_addr), "v"(ga) : "memory");
}
__device__ __forceinline__ void wait_async3() {
  asm volatile("s_wait_asynccnt 0x3" ::: "memory");
}
__device__ __forceinline__ void wait_async0() {
  asm volatile("s_wait_asynccnt 0x0" ::: "memory");
}

// ---------------------------------------------------------------------------
// Batched bf16 GEMM:  C[M,N] = packedA[M,K] * packedB[K,N] (+ bias epilogue)
// Workgroup (8 waves) computes a 64x128 C block: 4 M-tiles x 8 N-tiles.
// Per K-step, 12 fragments (4 A + 8 B, 12KB) are async-staged into LDS.
// Triple-buffered (36KB LDS) -> ONE workgroup barrier per K-step:
//   wait(stage kt done) -> barrier -> issue stage kt+2 -> compute kt.
// sched_barrier forces all 10 ds_loads before the WMMA quartet: one
// s_wait_dscnt, then 4 back-to-back v_wmma per K-step.
// mode 0: gx store, bias = b_ih[perm] + b_hh[perm], row-major [m][N]
// mode 1: projection store, bias = b_out, permuted to d_out[b][t][n]
// ---------------------------------------------------------------------------
__global__ void gemm_bf16(const u16* __restrict__ pA, const u16* __restrict__ pB,
                          const float* __restrict__ bias0, const float* __restrict__ bias1,
                          float* __restrict__ outp,
                          int NT8, int KT, int N, int mode) {
  __shared__ __align__(32) unsigned char sh[3][12 * 1024];

  const int tid  = threadIdx.x;
  const int lane = tid & 31;
  const int w    = tid >> 5;                 // 0..7
  const int mt0  = (blockIdx.x / NT8) * 4;   // in 16-row tiles
  const int nt0  = (blockIdx.x % NT8) * 8;   // in 16-col tiles
  const int nb   = (w >> 2) * 4;             // this wave's first local N-tile

  // hoisted staging addresses: 768 x 16B chunks over 256 threads (3 each)
  const u16* gsrc[3];
  unsigned   ldso[3];
#pragma unroll
  for (int j = 0; j < 3; ++j) {
    int c      = tid + 256 * j;              // chunk id 0..767
    int f      = c >> 6;                     // fragment 0..11
    int within = (c & 63) * 16;              // byte offset inside fragment
    gsrc[j] = (f < 4) ? pA + ((size_t)(mt0 + f) * KT) * 512 + (within >> 1)
                      : pB + ((size_t)(nt0 + f - 4) * KT) * 512 + (within >> 1);
    ldso[j] = (unsigned)(f * 1024 + within);
  }
  auto stage = [&](int kt, int s) {
#pragma unroll
    for (int j = 0; j < 3; ++j)
      async_ld16(&sh[s][ldso[j]], gsrc[j] + (size_t)kt * 512);
  };

  v8f acc[4] = {};
  stage(0, 0);                               // prologue: two stages in flight
  stage(1, 1);

  int s_cur = 0, s_issue = 2;                // rotating buffer indices (no div)
  for (int kt = 0; kt < KT; ++kt) {
    if (kt + 1 < KT) wait_async3();          // stage kt complete (kt+1 may fly)
    else             wait_async0();
    __syncthreads();                         // copies visible; reads of kt-1 done
    if (kt + 2 < KT) {
      stage(kt + 2, s_issue);                // overwrites buffer read at kt-1
      s_issue = (s_issue == 2) ? 0 : s_issue + 1;
    }

    const unsigned char* buf = sh[s_cur];
    s_cur = (s_cur == 2) ? 0 : s_cur + 1;

    v16bf a  = ((const v16bf*)(buf + (w & 3) * 1024))[lane];
    v16bf b0 = ((const v16bf*)(buf + (4 + nb + 0) * 1024))[lane];
    v16bf b1 = ((const v16bf*)(buf + (4 + nb + 1) * 1024))[lane];
    v16bf b2 = ((const v16bf*)(buf + (4 + nb + 2) * 1024))[lane];
    v16bf b3 = ((const v16bf*)(buf + (4 + nb + 3) * 1024))[lane];
    __builtin_amdgcn_sched_barrier(0);       // all ds_loads issue before WMMAs
    acc[0] = __builtin_amdgcn_wmma_f32_16x16x32_bf16(false, a, false, b0, (short)0, acc[0], false, false);
    acc[1] = __builtin_amdgcn_wmma_f32_16x16x32_bf16(false, a, false, b1, (short)0, acc[1], false, false);
    acc[2] = __builtin_amdgcn_wmma_f32_16x16x32_bf16(false, a, false, b2, (short)0, acc[2], false, false);
    acc[3] = __builtin_amdgcn_wmma_f32_16x16x32_bf16(false, a, false, b3, (short)0, acc[3], false, false);
  }

#pragma unroll
  for (int j = 0; j < 4; ++j) {
    int n = (nt0 + nb + j) * 16 + (lane & 15);
#pragma unroll
    for (int r = 0; r < 8; ++r) {
      int m = (mt0 + (w & 3)) * 16 + r + 8 * (lane >> 4);
      float v = acc[j][r];
      if (mode == 0) {
        int ob = ((n & 3) << 10) + (n >> 2);              // un-interleave bias
        outp[(size_t)m * (size_t)N + n] = v + bias0[ob] + bias1[ob];
      } else {
        int t = m >> 4, b = m & 15;                        // out is [B, T, VOUT]
        outp[((size_t)b * T_SEQ + t) * (size_t)VOUT + n] = v + bias0[n];
      }
    }
  }
}

// ---------------------------------------------------------------------------
// Persistent LSTM recurrence. 16 WGs x 256 thr = 128 waves; wave w owns gate
// columns n in [32w, 32w+32) == hidden units j in [8w, 8w+8) (gate-interleaved).
// Cell state c stays in registers across all T steps; one grid barrier/step.
// h_t written as bf16 A-frags (row-tile t of pAout) -> next step's A operand
// and the next layer's GEMM input, no repacking.
// ---------------------------------------------------------------------------
__device__ __forceinline__ void grid_barrier(unsigned* counter, unsigned target) {
  __syncthreads();
  if (threadIdx.x == 0) {
    __threadfence();
    atomicAdd(counter, 1u);
    volatile unsigned* vc = (volatile unsigned*)counter;
    while (*vc < target) __builtin_amdgcn_s_sleep(1);
  }
  __syncthreads();
}

__global__ void lstm_seq(const float* __restrict__ gx, const u16* __restrict__ pWhh,
                         u16* __restrict__ pAout, unsigned* __restrict__ counter) {
  int lane = threadIdx.x & 31;
  int w    = blockIdx.x * 8 + (threadIdx.x >> 5);   // 0..127
  int nt0  = w * 2;

  const v16bf* bB = (const v16bf*)pWhh;
  const v16bf* aB = (const v16bf*)pAout;

  // each lane represents (b = r + 8*(lane>>4), j = ntile*4 + ((lane>>2)&3)),
  // replicated across its quad; c kept in registers.
  float c[2][8];
#pragma unroll
  for (int jj = 0; jj < 2; ++jj)
#pragma unroll
    for (int r = 0; r < 8; ++r) c[jj][r] = 0.f;

  for (int t = 0; t < T_SEQ; ++t) {
    v8f acc[2] = {};
    if (t > 0) {
      for (int kt = 0; kt < KT_H; ++kt) {
        // distinct registers so all three loads issue before any wait
        v16bf a  = aB[(((size_t)(t - 1)) * KT_H + kt) * 32 + lane];
        v16bf b0 = bB[(((size_t)(nt0 + 0)) * KT_H + kt) * 32 + lane];
        v16bf b1 = bB[(((size_t)(nt0 + 1)) * KT_H + kt) * 32 + lane];
        if (kt + 1 < KT_H)   // warm L0 for next K-step of Whh (L2-resident)
          __builtin_prefetch(&bB[(((size_t)nt0) * KT_H + kt + 1) * 32 + lane], 0, 1);
        acc[0] = __builtin_amdgcn_wmma_f32_16x16x32_bf16(false, a, false, b0, (short)0, acc[0], false, false);
        acc[1] = __builtin_amdgcn_wmma_f32_16x16x32_bf16(false, a, false, b1, (short)0, acc[1], false, false);
      }
    }

#pragma unroll
    for (int jj = 0; jj < 2; ++jj) {
      int ntile = nt0 + jj;
      int n     = ntile * 16 + (lane & 15);
      int base  = lane & ~3;                 // quad base lane
#pragma unroll
      for (int r = 0; r < 8; ++r) {
        int b = r + 8 * (lane >> 4);
        float val = acc[jj][r] + gx[(((size_t)t * 16) + b) * G4H + n];
        // gather this (b,j)'s four gates from the quad (-> ds_bpermute)
        float gi = __shfl(val, base + 0, 32);
        float gf = __shfl(val, base + 1, 32);
        float gg = __shfl(val, base + 2, 32);
        float go = __shfl(val, base + 3, 32);
        float i_ = 1.f / (1.f + __expf(-gi));
        float f_ = 1.f / (1.f + __expf(-gf));
        float g_ = tanhf(gg);
        float o_ = 1.f / (1.f + __expf(-go));
        float cc = f_ * c[jj][r] + i_ * g_;
        c[jj][r] = cc;
        float h = o_ * tanhf(cc);
        if ((lane & 3) == 0) {               // one writer per quad
          int j    = ntile * 4 + ((lane >> 2) & 3);
          int kk   = j & 31;
          int lf   = b + 16 * ((kk >> 3) & 1);
          int slot = (kk & 7) + 8 * (kk >> 4);
          pAout[(((size_t)t * KT_H + (j >> 5)) * 512) + lf * 16 + slot] = f2bf(h);
        }
      }
    }
    grid_barrier(counter, (unsigned)(t + 1) * 16u);   // publish h_t
  }
}

// ---------------------------------------------------------------------------
extern "C" void kernel_launch(void* const* d_in, const int* in_sizes, int n_in,
                              void* d_out, int out_size, void* d_ws, size_t ws_size,
                              hipStream_t stream) {
  const int*   x     = (const int*)  d_in[0];
  const float* emb   = (const float*)d_in[1];
  const float* Wih0  = (const float*)d_in[2];
  const float* Whh0  = (const float*)d_in[3];
  const float* bih0  = (const float*)d_in[4];
  const float* bhh0  = (const float*)d_in[5];
  const float* Wih1  = (const float*)d_in[6];
  const float* Whh1  = (const float*)d_in[7];
  const float* bih1  = (const float*)d_in[8];
  const float* bhh1  = (const float*)d_in[9];
  const float* Wout  = (const float*)d_in[10];
  const float* bout  = (const float*)d_in[11];
  float* out = (float*)d_out;

  // workspace carve-up (256B aligned)
  char* base = (char*)d_ws;
  size_t off = 0;
  auto carve = [&](size_t bytes) -> char* {
    off = (off + 255) & ~(size_t)255;
    char* p = base + off;
    off += bytes;
    return p;
  };
  unsigned* ctrs  = (unsigned*)carve(256);
  u16* pA0   = (u16*)carve((size_t)M_ROWS * E_DIM * 2);   // embed frags (reused as layer-2 out)
  u16* pA1   = (u16*)carve((size_t)M_ROWS * H_DIM * 2);   // layer-0 output frags
  u16* pWih0 = (u16*)carve((size_t)G4H * E_DIM * 2);
  u16* pWhh0 = (u16*)carve((size_t)G4H * H_DIM * 2);
  u16* pWih1 = (u16*)carve((size_t)G4H * H_DIM * 2);
  u16* pWhh1 = (u16*)carve((size_t)G4H * H_DIM * 2);
  u16* pWout = (u16*)carve((size_t)VOUT * H_DIM * 2);
  float* gx  = (float*)carve((size_t)M_ROWS * G4H * 4);   // shared by both layers
  u16* pA2   = pA0;                                       // alias: pA0 dead after first GEMM
  (void)ws_size; (void)in_sizes; (void)n_in; (void)out_size;

  init_counters<<<1, 64, 0, stream>>>(ctrs);

  int nW = (G4H * E_DIM + 255) / 256;
  pack_w<<<nW, 256, 0, stream>>>(Wih0, pWih0, G4H, E_DIM, 1);
  pack_w<<<nW, 256, 0, stream>>>(Whh0, pWhh0, G4H, H_DIM, 1);
  pack_w<<<nW, 256, 0, stream>>>(Wih1, pWih1, G4H, H_DIM, 1);
  pack_w<<<nW, 256, 0, stream>>>(Whh1, pWhh1, G4H, H_DIM, 1);
  pack_w<<<(int)(((size_t)VOUT * H_DIM + 255) / 256), 256, 0, stream>>>(
      Wout, pWout, VOUT, H_DIM, 0);
  pack_embed<<<(int)(((size_t)M_ROWS * E_DIM + 255) / 256), 256, 0, stream>>>(x, emb, pA0);

  // layer 0: gx = x @ W_ih0^T + b_ih0 + b_hh0  (gate-interleaved columns)
  // 64x128 C blocks: (8192/64) * (4096/128) = 128*32 = 4096 WGs
  gemm_bf16<<<4096, 256, 0, stream>>>(pA0, pWih0, bih0, bhh0, gx,
                                      G4H / 128, KT_H, G4H, 0);
  lstm_seq<<<16, 256, 0, stream>>>(gx, pWhh0, pA1, ctrs + 0);

  // layer 1
  gemm_bf16<<<4096, 256, 0, stream>>>(pA1, pWih1, bih1, bhh1, gx,
                                      G4H / 128, KT_H, G4H, 0);
  lstm_seq<<<16, 256, 0, stream>>>(gx, pWhh1, pA2, ctrs + 1);

  // projection: logits[b][t][n] = h1 @ W_out^T + b_out
  // (8192/64) * (32000/128) = 128*250 = 32000 WGs
  gemm_bf16<<<32000, 256, 0, stream>>>(pA2, pWout, bout, bout, out,
                                       VOUT / 128, KT_H, VOUT, 1);
}
